// CrossCameraTripletLoss_66967130079564
// MI455X (gfx1250) — matrix operand order repaired and checked by reading, not compiled, not run on hardware
//
#include <hip/hip_runtime.h>
#include <hip/hip_bf16.h>
#include <float.h>

typedef __attribute__((ext_vector_type(16))) __bf16 v16bf;
typedef __attribute__((ext_vector_type(8)))  __bf16 v8bf;
typedef __attribute__((ext_vector_type(8)))  float  v8f;

#define N 8192
#define D 128
#define MARGIN 0.2f
#define EPSC 1e-6f

static __device__ __forceinline__ __bf16 bits2bf(unsigned short u) {
    return __builtin_bit_cast(__bf16, u);
}

// ---------------------------------------------------------------------------
// Kernel 1: split f32 features into bf16 hi/lo planes + row squared norms.
// One wave per row; each lane handles 4 contiguous elements.
// ---------------------------------------------------------------------------
__global__ void __launch_bounds__(256)
prep_kernel(const float* __restrict__ f, float* __restrict__ sq,
            __bf16* __restrict__ hi, __bf16* __restrict__ lo) {
    const int row  = (blockIdx.x * blockDim.x + threadIdx.x) >> 5;
    const int lane = threadIdx.x & 31;
    if (row >= N) return;
    const float* src = f + (size_t)row * D;
    float acc = 0.f;
#pragma unroll
    for (int e = 0; e < 4; ++e) {
        const int k = lane * 4 + e;
        const float x = src[k];
        const unsigned u = __float_as_uint(x);
        const unsigned short hb = (unsigned short)(u >> 16);        // truncated bf16 hi
        const float hf = __uint_as_float(u & 0xFFFF0000u);          // exact in f32
        const float lf = x - hf;                                    // exact residual
        const unsigned short lb = (unsigned short)(__float_as_uint(lf) >> 16);
        hi[(size_t)row * D + k] = bits2bf(hb);
        lo[(size_t)row * D + k] = bits2bf(lb);
        acc = fmaf(x, x, acc);
    }
#pragma unroll
    for (int m = 16; m >= 1; m >>= 1) acc += __shfl_xor(acc, m, 32);
    if (lane == 0) sq[row] = acc;
}

// ---------------------------------------------------------------------------
// Kernel 2: fused Gram + batch-hard mining.
// One wave per 16-anchor row block. Per 16-col tile:
//   G(16x16) via 12 x v_wmma_f32_16x16x32_bf16 (hi*hi + hi*lo + lo*hi),
// all 16 B-plane loads hoisted ahead of the WMMA block so they issue as one
// clause and overlap the matrix ops. Mining runs on t = G - 0.5*sq[j]
// (dist^2 = sq[i] - 2t): hardest positive = min t, hardest negative = max t.
// Branch-free cndmask updates; strict compares keep first-occurrence
// (lowest j) per lane; index tie-break only in the cross-lane reduce.
// ---------------------------------------------------------------------------
__global__ void __launch_bounds__(128)
mine_kernel(const __bf16* __restrict__ hi, const __bf16* __restrict__ lo,
            const float* __restrict__ sq, const int* __restrict__ labels,
            int* __restrict__ pos_idx, int* __restrict__ neg_idx,
            float* __restrict__ validf) {
    const int lane   = threadIdx.x & 31;
    const int wave   = threadIdx.x >> 5;
    const int rowblk = blockIdx.x * 4 + wave;
    const int ibase  = rowblk * 16;
    const int half   = lane >> 4;
    const int colsub = lane & 15;

    // --- A tile (16 rows x 128 K), bf16 16x32 A layout:
    //     lane l holds row ibase+(l&15); K offsets {16h..16h+15}
    const size_t arow = (size_t)(ibase + colsub) * D;
    v16bf Ah[4], Al[4];
#pragma unroll
    for (int c = 0; c < 4; ++c) {
        const int k0 = c * 32 + half * 16;
        const int k1 = c * 32 + half * 16 + 8;
        v8bf h0 = *(const v8bf*)(hi + arow + k0);
        v8bf h1 = *(const v8bf*)(hi + arow + k1);
        v8bf l0 = *(const v8bf*)(lo + arow + k0);
        v8bf l1 = *(const v8bf*)(lo + arow + k1);
        Ah[c] = __builtin_shufflevector(h0, h1, 0,1,2,3,4,5,6,7,8,9,10,11,12,13,14,15);
        Al[c] = __builtin_shufflevector(l0, l1, 0,1,2,3,4,5,6,7,8,9,10,11,12,13,14,15);
    }

    int labi[8];
#pragma unroll
    for (int r = 0; r < 8; ++r) labi[r] = labels[ibase + r + half * 8];

    float pv[8], nv[8];
    int   pi[8], ni[8];
#pragma unroll
    for (int r = 0; r < 8; ++r) {
        pv[r] = FLT_MAX;   // minimizing t for positives
        nv[r] = -FLT_MAX;  // maximizing t for negatives
        pi[r] = 0x7fffffff; ni[r] = 0x7fffffff;
    }

    for (int jb = 0; jb < N; jb += 16) {
        const int   j    = jb + colsub;
        const float sqj  = sq[j];
        const int   labj = labels[j];
        const size_t brow = (size_t)j * D;

        // Warm the WGP-level caches for the next tile (clamped: branch-free).
        const int jn = (jb + 16 < N) ? (jb + 16) : jb;
        const size_t prow = (size_t)(jn + colsub) * D;
        __builtin_prefetch(hi + prow, 0, 3);
        __builtin_prefetch(lo + prow, 0, 3);

        // Hoist all B-plane loads for this tile: one clause of 16 b128 loads.
        // bf16 32x16 B layout: lane l holds col jb+(l&15), K = 32c + 16h + e
        v16bf Bh[4], Bl[4];
#pragma unroll
        for (int c = 0; c < 4; ++c) {
            const int kb = c * 32 + half * 16;
            Bh[c] = *(const v16bf*)(hi + brow + kb);
            Bl[c] = *(const v16bf*)(lo + brow + kb);
        }

        v8f a0 = {}, a1 = {}, a2 = {};
#pragma unroll
        for (int c = 0; c < 4; ++c) {
            a0 = __builtin_amdgcn_wmma_f32_16x16x32_bf16(false, Ah[c], false, Bh[c],
                                                         (short)0, a0, false, false);
            a1 = __builtin_amdgcn_wmma_f32_16x16x32_bf16(false, Ah[c], false, Bl[c],
                                                         (short)0, a1, false, false);
            a2 = __builtin_amdgcn_wmma_f32_16x16x32_bf16(false, Al[c], false, Bh[c],
                                                         (short)0, a2, false, false);
        }

        const float bias = -0.5f * sqj;           // per-column bias, per-lane const
#pragma unroll
        for (int r = 0; r < 8; ++r) {
            const float t = (a0[r] + a1[r]) + (a2[r] + bias);   // G - 0.5*sq[j]
            const bool isPos = (labj == labi[r]);
            const bool bp = isPos && (t < pv[r]);        // hardest positive: min t
            pv[r] = bp ? t : pv[r];
            pi[r] = bp ? j : pi[r];
            const bool bn = (!isPos) && (t > nv[r]);     // hardest negative: max t
            nv[r] = bn ? t : nv[r];
            ni[r] = bn ? j : ni[r];
        }
    }

    // Reduce across the 16 lanes of each half-group (xor stays within group).
#pragma unroll
    for (int m = 1; m <= 8; m <<= 1) {
#pragma unroll
        for (int r = 0; r < 8; ++r) {
            float opv = __shfl_xor(pv[r], m, 32);
            int   opi = __shfl_xor(pi[r], m, 32);
            bool tp = (opv < pv[r]) || (opv == pv[r] && opi < pi[r]);
            pv[r] = tp ? opv : pv[r];
            pi[r] = tp ? opi : pi[r];
            float onv = __shfl_xor(nv[r], m, 32);
            int   oni = __shfl_xor(ni[r], m, 32);
            bool tn = (onv > nv[r]) || (onv == nv[r] && oni < ni[r]);
            nv[r] = tn ? onv : nv[r];
            ni[r] = tn ? oni : ni[r];
        }
    }

    if (colsub == 0) {
#pragma unroll
        for (int r = 0; r < 8; ++r) {
            const int i = ibase + r + half * 8;
            const int ok = (nv[r] > -FLT_MAX) ? 1 : 0;
            pos_idx[i] = pi[r];
            neg_idx[i] = ok ? ni[r] : 0;
            validf[i]  = (float)ok;
        }
    }
}

// ---------------------------------------------------------------------------
// Kernel 3: per-anchor hinge in exact f32 on the original features.
// One wave per anchor; 4 dims per lane.
// ---------------------------------------------------------------------------
__global__ void __launch_bounds__(256)
loss_kernel(const float* __restrict__ f, const int* __restrict__ pos_idx,
            const int* __restrict__ neg_idx, const float* __restrict__ validf,
            float* __restrict__ per) {
    const int i    = (blockIdx.x * blockDim.x + threadIdx.x) >> 5;
    const int lane = threadIdx.x & 31;
    if (i >= N) return;
    const int p = pos_idx[i];
    const int q = neg_idx[i];
    const float v = validf[i];
    const float* A = f + (size_t)i * D;
    const float* P = f + (size_t)p * D;
    const float* Q = f + (size_t)q * D;
    float sp = 0.f, sn = 0.f;
#pragma unroll
    for (int e = 0; e < 4; ++e) {
        const int k = lane * 4 + e;
        const float dp = A[k] - P[k] + EPSC;
        const float dn = A[k] - Q[k] + EPSC;
        sp = fmaf(dp, dp, sp);
        sn = fmaf(dn, dn, sn);
    }
#pragma unroll
    for (int m = 16; m >= 1; m >>= 1) {
        sp += __shfl_xor(sp, m, 32);
        sn += __shfl_xor(sn, m, 32);
    }
    if (lane == 0)
        per[i] = (v > 0.f) ? fmaxf(0.f, sqrtf(sp) - sqrtf(sn) + MARGIN) : 0.f;
}

// ---------------------------------------------------------------------------
// Kernel 4: deterministic single-block reduction -> scalar loss.
// ---------------------------------------------------------------------------
__global__ void __launch_bounds__(256)
reduce_kernel(const float* __restrict__ per, const float* __restrict__ validf,
              float* __restrict__ out) {
    __shared__ float ssum[256];
    __shared__ float scnt[256];
    float s = 0.f, c = 0.f;
    for (int i = threadIdx.x; i < N; i += 256) { s += per[i]; c += validf[i]; }
    ssum[threadIdx.x] = s;
    scnt[threadIdx.x] = c;
    __syncthreads();
    for (int st = 128; st > 0; st >>= 1) {
        if (threadIdx.x < st) {
            ssum[threadIdx.x] += ssum[threadIdx.x + st];
            scnt[threadIdx.x] += scnt[threadIdx.x + st];
        }
        __syncthreads();
    }
    if (threadIdx.x == 0) out[0] = (scnt[0] > 0.f) ? ssum[0] / scnt[0] : 0.f;
}

// ---------------------------------------------------------------------------
extern "C" void kernel_launch(void* const* d_in, const int* in_sizes, int n_in,
                              void* d_out, int out_size, void* d_ws, size_t ws_size,
                              hipStream_t stream) {
    const float* feat   = (const float*)d_in[0];
    const int*   labels = (const int*)d_in[1];
    float*       out    = (float*)d_out;
    (void)in_sizes; (void)n_in; (void)out_size; (void)ws_size;

    char* ws = (char*)d_ws;
    float* sq      = (float*)(ws + 0);            // 8192 f32
    int*   pos_idx = (int*)  (ws + 32768);        // 8192 i32
    int*   neg_idx = (int*)  (ws + 65536);        // 8192 i32
    float* validf  = (float*)(ws + 98304);        // 8192 f32
    float* per     = (float*)(ws + 131072);       // 8192 f32
    __bf16* hi     = (__bf16*)(ws + 262144);                          // N*D bf16 (2 MB)
    __bf16* lo     = (__bf16*)(ws + 262144 + (size_t)N * D * 2);      // N*D bf16 (2 MB)

    prep_kernel  <<<N / 8,      256, 0, stream>>>(feat, sq, hi, lo);
    mine_kernel  <<<N / 16 / 4, 128, 0, stream>>>(hi, lo, sq, labels,
                                                  pos_idx, neg_idx, validf);
    loss_kernel  <<<N / 8,      256, 0, stream>>>(feat, pos_idx, neg_idx, validf, per);
    reduce_kernel<<<1,          256, 0, stream>>>(per, validf, out);
}
